// EMAX_60756607369740
// MI455X (gfx1250) — matrix operand order repaired and checked by reading, not compiled, not run on hardware
//
#include <hip/hip_runtime.h>
#include <hip/hip_bf16.h>

typedef __attribute__((ext_vector_type(16))) _Float16 v16h;
typedef __attribute__((ext_vector_type(8)))  float    v8f;
typedef __attribute__((ext_vector_type(4)))  int      v4i;

#define BB 16
#define CC_ 512
#define HH 64
#define WW 64
#define GG 32
#define CG 16

// fast sigmoid: v_rcp_f32 instead of the IEEE division expansion
__device__ __forceinline__ float sigf(float x) {
#if __has_builtin(__builtin_amdgcn_rcpf)
    return __builtin_amdgcn_rcpf(1.0f + __expf(-x));
#else
    return 1.0f / (1.0f + __expf(-x));
#endif
}

// ---- CDNA5 async global->LDS copy (prototype revealed by compiler probe:
//      (addrspace(1) v4i*, addrspace(3) v4i*, imm offset, imm cpol)) --------
#if __has_builtin(__builtin_amdgcn_global_load_async_to_lds_b128)
#define HAVE_ASYNC_LDS 1
__device__ __forceinline__ void async_copy16(void* lds, const void* g) {
    __builtin_amdgcn_global_load_async_to_lds_b128(
        (__attribute__((address_space(1))) v4i*)g,
        (__attribute__((address_space(3))) v4i*)lds, 0, 0);
}
__device__ __forceinline__ void async_wait0() {
#if __has_builtin(__builtin_amdgcn_s_wait_asynccnt)
    __builtin_amdgcn_s_wait_asynccnt(0);
#else
    asm volatile("s_wait_asynccnt 0" ::: "memory");
#endif
}
#endif

// ---------------------------------------------------------------------------
// K1: global average pool of x per (b,c)  ->  p[b*512+c] (mean, /4096 applied)
// ---------------------------------------------------------------------------
__global__ void pool_kernel(const float* __restrict__ x, float* __restrict__ p) {
    const int bc = blockIdx.x;                 // 0..8191
    const float* src = x + (size_t)bc * 4096;
    float s = 0.0f;
#if defined(HAVE_ASYNC_LDS)
    __shared__ float stage[4096];
    #pragma unroll
    for (int it = 0; it < 4; ++it) {
        const int i = (it * 256 + threadIdx.x) * 4;
        async_copy16(&stage[i], src + i);
    }
    async_wait0();
    __syncthreads();
    for (int i = threadIdx.x; i < 4096; i += 256) s += stage[i];
#else
    for (int i = threadIdx.x; i < 4096; i += 256) {
        __builtin_prefetch(src + i + 1024, 0, 0);
        s += src[i];
    }
#endif
    __shared__ float red[256];
    red[threadIdx.x] = s;
    __syncthreads();
    for (int off = 128; off > 0; off >>= 1) {
        if (threadIdx.x < off) red[threadIdx.x] += red[threadIdx.x + off];
        __syncthreads();
    }
    if (threadIdx.x == 0) p[bc] = red[0] * (1.0f / 4096.0f);
}

// ---------------------------------------------------------------------------
// K2/K4: squeeze-excite FC stack: sig( W2 @ relu(W1 @ p + b1) + b2 )
// ---------------------------------------------------------------------------
__global__ void se_fc_kernel(const float* __restrict__ pin,
                             const float* __restrict__ wa, const float* __restrict__ ba,
                             const float* __restrict__ wb, const float* __restrict__ bb,
                             float* __restrict__ sout, float scale) {
    const int b = blockIdx.x;
    __shared__ float pv[512];
    __shared__ float hd[32];
    for (int i = threadIdx.x; i < 512; i += blockDim.x) pv[i] = pin[b * 512 + i] * scale;
    __syncthreads();
    if (threadIdx.x < 32) {
        float a = ba[threadIdx.x];
        const float* wr = wa + threadIdx.x * 512;
        for (int k = 0; k < 512; ++k) a += wr[k] * pv[k];
        hd[threadIdx.x] = a > 0.0f ? a : 0.0f;
    }
    __syncthreads();
    for (int c = threadIdx.x; c < 512; c += blockDim.x) {
        float a = bb[c];
        const float* wr = wb + c * 32;
        #pragma unroll
        for (int k = 0; k < 32; ++k) a += wr[k] * hd[k];
        sout[b * 512 + c] = sigf(a);
    }
}

// ---------------------------------------------------------------------------
// K3: main fused kernel — one block per group (bg = b*32+g), 512 threads.
// gx staged in LDS as f16, layout [y][x][ic] (ic fastest, 32B per pixel =>
// one aligned v16h per (y,x)). Coordinate gates live in cgate[c][0..127]
// (xh | xw concatenated; per-channel stride 132 => conflict-free banks).
// ---------------------------------------------------------------------------
#define GSTRIDE 132

__global__ void __launch_bounds__(512) main_kernel(
    const float* __restrict__ x,
    const float* __restrict__ w1, const float* __restrict__ b1,
    const float* __restrict__ w3, const float* __restrict__ b3,
    const float* __restrict__ gn_w, const float* __restrict__ gn_b,
    const float* __restrict__ s1ws, const float* __restrict__ gamma,
    float* __restrict__ out, float* __restrict__ pool2)
{
    extern __shared__ v16h smem_v[];
    _Float16* gxh = (_Float16*)smem_v;                       // 64*64*16 halves = 128KB
    float* fbase = (float*)((char*)smem_v + 131072);
    float* cgate = fbase;              // 16 * 132  (xh gates @ [c][0..63], xw @ [c][64..127])
    float* sumb  = fbase + 2112;       // 16
    float* sqb   = fbase + 2128;       // 16
    float* x2s   = fbase + 2144;       // 16
    float* fsb   = fbase + 2160;       // 16
    float* mu    = fbase + 2176;       // 16
    float* k1c   = fbase + 2192;       // 16
    float* a11   = fbase + 2208;       // 16
    float* a21   = fbase + 2224;       // 16
    float* b3c   = fbase + 2240;       // 16
    float* gnbc  = fbase + 2256;       // 16
    float* s1r   = fbase + 2272;       // 16
    float* gmm   = fbase + 2288;       // 1

    const int tid  = threadIdx.x;
    const int lane = tid & 31;
    const int wid  = tid >> 5;          // wave id 0..15 (wave32)
    const int half = lane >> 4;         // 0: lanes 0-15, 1: lanes 16-31
    const int nn   = lane & 15;
    const int bg = blockIdx.x;
    const int b  = bg >> 5;
    const int g  = bg & 31;
    const float* gx = x + ((size_t)(b * 512 + g * 16)) * 4096;   // [ic][y][x]

    // ---- stage f32 -> f16 LDS (transposing to ic-fastest) ---------------
    for (int i4 = tid; i4 < 16384; i4 += 512) {
        const int idx = i4 << 2;
        const int ic  = idx >> 12;
        const int rem = idx & 4095;      // y*64 + x (multiple of 4)
        const float4 v = *(const float4*)(gx + (size_t)ic * 4096 + rem);
        gxh[(rem + 0) * 16 + ic] = (_Float16)v.x;
        gxh[(rem + 1) * 16 + ic] = (_Float16)v.y;
        gxh[(rem + 2) * 16 + ic] = (_Float16)v.z;
        gxh[(rem + 3) * 16 + ic] = (_Float16)v.w;
    }
    __syncthreads();

    // ---- xh / xw means into cgate (branchless: slot = c*GSTRIDE + l) ----
    for (int i = tid; i < 2048; i += 512) {
        const int c   = i & 15;
        const int l   = (i >> 4) & 63;
        const int isw = i >> 10;
        float s = 0.0f;
        if (!isw) {
            for (int xx = 0; xx < 64; ++xx) s += (float)gxh[(((l << 6) + xx) << 4) + c];
        } else {
            for (int yy = 0; yy < 64; ++yy) s += (float)gxh[(((yy << 6) + l) << 4) + c];
        }
        cgate[c * GSTRIDE + isw * 64 + l] = s * (1.0f / 64.0f);
    }
    if (tid < 16) { sumb[tid] = 0.0f; sqb[tid] = 0.0f; x2s[tid] = 0.0f; fsb[tid] = 0.0f; }
    __syncthreads();

    // ---- coordinate matmul hw = w1 @ [xh;xw], via WMMA (waves 0-7) ------
    if (wid < 8) {
        const int l0 = wid << 4;
        const int l  = l0 + nn;
        v16h af = {};
        #pragma unroll
        for (int h = 0; h < 16; ++h) {
            const int k = (half << 4) + h;   // K in A layout
            if (k < 16) af[h] = (_Float16)w1[nn * 16 + k];          // M = nn
        }
        v16h bf = {};
        if (!half) {                                                // K = h (c); K>=16 zero pad
            #pragma unroll
            for (int h = 0; h < 16; ++h)
                bf[h] = (_Float16)cgate[h * GSTRIDE + l];           // cat[c][l] direct
        }
        v8f cf = {};
        cf = __builtin_amdgcn_wmma_f32_16x16x32_f16(false, af, false, bf, (short)0, cf, false, false);
        #pragma unroll
        for (int v = 0; v < 8; ++v) {
            const int m = v + (half << 3);
            cgate[m * GSTRIDE + l] = sigf(cf[v] + b1[m]);           // branchless store
        }
    }
    __syncthreads();

    // ---- conv A-fragments from w3 (shared across all groups) ------------
    v16h afr[5];
    {
        #pragma unroll
        for (int cc = 0; cc < 5; ++cc) {
            v16h a = {};
            #pragma unroll
            for (int h = 0; h < 16; ++h) {
                const int tap = 2 * cc + half;
                const int icc = h;
                if (tap < 9) a[h] = (_Float16)w3[(nn * 16 + icc) * 9 + tap];
            }
            afr[cc] = a;
        }
    }

    // B-fragment loader: tap = 2*cc + half; ic = h (contiguous 32B in LDS)
    auto bload = [&](int y, int xb, int cc) -> v16h {
        const int tap = 2 * cc + half;
        v16h bv = {};
        if (tap < 9) {
            const int dy  = tap / 3 - 1;
            const int dx  = tap - (tap / 3) * 3 - 1;
            const int row = y + dy;
            const int col = (xb << 4) + nn + dx;
            if ((unsigned)row < 64u && (unsigned)col < 64u)
                bv = *(const v16h*)(gxh + ((((row << 6) + col)) << 4));
        }
        return bv;
    };

    // ---- conv pass 1: accumulate x2 channel sums ------------------------
    {
        v8f acc = {};
        for (int k = 0; k < 4; ++k) {
            const int y = wid + (k << 4);
            #pragma unroll
            for (int xb = 0; xb < 4; ++xb) {
                #pragma unroll
                for (int cc = 0; cc < 5; ++cc) {
                    v16h bf = bload(y, xb, cc);
                    acc = __builtin_amdgcn_wmma_f32_16x16x32_f16(false, afr[cc], false, bf,
                                                                 (short)0, acc, false, false);
                }
            }
        }
        #pragma unroll
        for (int v = 0; v < 8; ++v) {
            float t = acc[v];
            t += __shfl_xor(t, 1); t += __shfl_xor(t, 2);
            t += __shfl_xor(t, 4); t += __shfl_xor(t, 8);
            if ((lane & 15) == 0) atomicAdd(&x2s[(half << 3) + v], t);
        }
    }

    // ---- x1 statistics (mean / var per channel) -------------------------
    {
        const int c = tid & 15;
        const int p0 = (tid >> 4) << 7;
        float s = 0.0f, ss = 0.0f;
        for (int p = p0; p < p0 + 128; ++p) {
            const int yy = p >> 6, xx = p & 63;
            const float gv = (float)gxh[(p << 4) + c];
            const float x1 = gv * cgate[c * GSTRIDE + yy] * cgate[c * GSTRIDE + 64 + xx];
            s += x1; ss += x1 * x1;
        }
        atomicAdd(&sumb[c], s);
        atomicAdd(&sqb[c], ss);
    }
    __syncthreads();

    // ---- finalize stats, softmaxes, scalars (thread 0) ------------------
    if (tid == 0) {
        float a21v[16], a11v[16];
        float mx2 = -1e30f, mx1 = -1e30f;
        for (int c = 0; c < 16; ++c) {
            const float m_ = sumb[c] * (1.0f / 4096.0f);
            const float v_ = sqb[c] * (1.0f / 4096.0f) - m_ * m_;
            mu[c]   = m_;
            k1c[c]  = rsqrtf(v_ + 1e-5f) * gn_w[c];
            gnbc[c] = gn_b[c];
            b3c[c]  = b3[c];
            s1r[c]  = s1ws[b * 512 + g * 16 + c];
            const float x2m = x2s[c] * (1.0f / 4096.0f) + b3[c];
            a21v[c] = x2m; if (x2m > mx2) mx2 = x2m;
            // post-GN spatial mean of x1 is exactly gn_b[c]
            a11v[c] = gn_b[c]; if (gn_b[c] > mx1) mx1 = gn_b[c];
        }
        float d2 = 0.0f, d1 = 0.0f;
        for (int c = 0; c < 16; ++c) {
            a21v[c] = __expf(a21v[c] - mx2); d2 += a21v[c];
            a11v[c] = __expf(a11v[c] - mx1); d1 += a11v[c];
        }
        for (int c = 0; c < 16; ++c) { a21[c] = a21v[c] / d2; a11[c] = a11v[c] / d1; }
        gmm[0] = gamma[0];
    }
    __syncthreads();

    // ---- conv pass 2 fused: wv, spatial, channel, fused store + pool2 ---
    {
        float facc[8] = {0, 0, 0, 0, 0, 0, 0, 0};
        const float gm = gmm[0];
        for (int k = 0; k < 4; ++k) {
            const int y = wid + (k << 4);
            #pragma unroll
            for (int xb = 0; xb < 4; ++xb) {
                v8f cfr = {};
                #pragma unroll
                for (int cc = 0; cc < 5; ++cc) {
                    v16h bf = bload(y, xb, cc);
                    cfr = __builtin_amdgcn_wmma_f32_16x16x32_f16(false, afr[cc], false, bf,
                                                                 (short)0, cfr, false, false);
                }
                const int xx = (xb << 4) + nn;
                float wvv = 0.0f;
                #pragma unroll
                for (int v = 0; v < 8; ++v) {
                    const int m = v + (half << 3);
                    wvv += a11[m] * (cfr[v] + b3c[m]);       // a11 . x2
                }
                const _Float16* hp = gxh + ((((y << 6) + xx)) << 4);
                float gvs[8];
                #pragma unroll
                for (int j = 0; j < 8; ++j) {
                    const int c = j + (half << 3);
                    const float gv = (float)hp[c];
                    gvs[j] = gv;
                    const float x1 = gv * cgate[c * GSTRIDE + y] * cgate[c * GSTRIDE + 64 + xx];
                    const float xg = (x1 - mu[c]) * k1c[c] + gnbc[c];
                    wvv += a21[c] * xg;                       // a21 . x1gn
                }
                wvv += __shfl_xor(wvv, 16);                   // combine channel halves
                const float sg = sigf(wvv);
                #pragma unroll
                for (int j = 0; j < 8; ++j) {
                    const int c = j + (half << 3);
                    const float gv = gvs[j];
                    const float fused = gv * sg + gm * gv * s1r[c];
                    out[(((size_t)(b * 512 + g * 16 + c)) << 12) + (y << 6) + xx] = fused;
                    facc[j] += fused;
                }
            }
        }
        #pragma unroll
        for (int j = 0; j < 8; ++j) atomicAdd(&fsb[j + (half << 3)], facc[j]);
    }
    __syncthreads();
    if (tid < 16) pool2[bg * 16 + tid] = fsb[tid];   // raw sum, /4096 in SE2
}

// ---------------------------------------------------------------------------
// K5: out *= s2[b,c]  (in place on d_out)
// ---------------------------------------------------------------------------
__global__ void apply_kernel(float* __restrict__ out, const float* __restrict__ s2) {
    const size_t total  = (size_t)BB * CC_ * HH * WW;
    const size_t stride = (size_t)gridDim.x * blockDim.x;
    for (size_t i = (size_t)blockIdx.x * blockDim.x + threadIdx.x; i < total; i += stride) {
        const int bc = (int)(i >> 12);
        out[i] *= s2[bc];
    }
}

// ---------------------------------------------------------------------------
extern "C" void kernel_launch(void* const* d_in, const int* in_sizes, int n_in,
                              void* d_out, int out_size, void* d_ws, size_t ws_size,
                              hipStream_t stream) {
    const float* x     = (const float*)d_in[0];
    const float* w1    = (const float*)d_in[1];
    const float* b1    = (const float*)d_in[2];
    const float* w3    = (const float*)d_in[3];
    const float* b3    = (const float*)d_in[4];
    const float* gn_w  = (const float*)d_in[5];
    const float* gn_b  = (const float*)d_in[6];
    const float* cg_w1 = (const float*)d_in[7];
    const float* cg_b1 = (const float*)d_in[8];
    const float* cg_w2 = (const float*)d_in[9];
    const float* cg_b2 = (const float*)d_in[10];
    const float* ga_w1 = (const float*)d_in[11];
    const float* ga_b1 = (const float*)d_in[12];
    const float* ga_w2 = (const float*)d_in[13];
    const float* ga_b2 = (const float*)d_in[14];
    const float* gamma = (const float*)d_in[15];

    float* ws    = (float*)d_ws;
    float* p     = ws;               // 8192
    float* s1    = ws + 8192;        // 8192
    float* pool2 = ws + 16384;       // 8192
    float* s2    = ws + 24576;       // 8192

    pool_kernel<<<BB * CC_, 256, 0, stream>>>(x, p);
    se_fc_kernel<<<BB, 256, 0, stream>>>(p, cg_w1, cg_b1, cg_w2, cg_b2, s1, 1.0f);

    const size_t smem = 131072 + 10240;   // f16 tile + f32 scratch (cgate etc.)
    main_kernel<<<BB * GG, 512, smem, stream>>>(x, w1, b1, w3, b3, gn_w, gn_b,
                                                s1, gamma, (float*)d_out, pool2);

    se_fc_kernel<<<BB, 256, 0, stream>>>(pool2, ga_w1, ga_b1, ga_w2, ga_b2, s2,
                                         1.0f / 4096.0f);
    apply_kernel<<<4096, 256, 0, stream>>>((float*)d_out, s2);
}
